// Bond2AtomLayer_5119601016922
// MI455X (gfx1250) — matrix operand order repaired and checked by reading, not compiled, not run on hardware
//
#include <hip/hip_runtime.h>
#include <hip/hip_bf16.h>
#include <math.h>

#define HID   128
#define NHEAD 8
#define DHEAD 16

typedef __attribute__((ext_vector_type(16))) __bf16 v16bf;
typedef __attribute__((ext_vector_type(8)))  __bf16 v8bf;
typedef __attribute__((ext_vector_type(8)))  float  v8f;

// ---------------------------------------------------------------------------
// Float atomic max via sign-aware integer atomics (standard correct trick).
// ---------------------------------------------------------------------------
__device__ __forceinline__ void atomic_max_float(float* addr, float value) {
    if (value >= 0.0f) {
        atomicMax((int*)addr, __float_as_int(value));
    } else {
        atomicMin((unsigned int*)addr, __float_as_uint(value));
    }
}

__device__ __forceinline__ float lrelu(float v) {
    return (v > 0.0f) ? v : 0.01f * v;   // leaky_relu slope 0.01
}

// ---------------------------------------------------------------------------
// One-time f32 -> bf16 weight conversion (weights then live in L2 at half
// the bytes; removes per-block cvt work from the GEMM hot loop).
// ---------------------------------------------------------------------------
__global__ void convert_w_kernel(const float* __restrict__ w,
                                 __bf16* __restrict__ out)
{
    int idx = blockIdx.x * blockDim.x + threadIdx.x;   // 0..16383
    out[idx] = (__bf16)w[idx];
}

// ---------------------------------------------------------------------------
// Y[rows,128] = act(X[rows,128] @ W^T + bias) (+ residual), W pre-converted
// to bf16, [out,in] row-major. One 256-thread block = 8 waves; wave w owns
// the 16x16 tile at column 16*w of a 16-row strip.
// X tile converted to bf16 once at LDS-fill time; K loop is pure
// ds_load_b128 / global_load_b128 -> 4 x v_wmma_f32_16x16x32_bf16.
// LDS row stride = 136 bf16 (272B = 68 banks) -> A-fragment rows hit
// disjoint bank quads. Epilogue: wave-uniform full-tile fast path (no
// per-row EXEC manipulation); RELU/HAS_RES are compile-time.
// ---------------------------------------------------------------------------
template <bool RELU, bool HAS_RES>
__global__ __launch_bounds__(256) void gemm_bias_act_kernel(
    const float*  __restrict__ X,
    const __bf16* __restrict__ Wbf,
    const float*  __restrict__ bias,
    const float*  res,                 // may alias Y (residual-into-self)
    float* Y,
    int rows)
{
    __shared__ __bf16 Xs[16][HID + 8];

    const int tid  = threadIdx.x;
    const int row0 = blockIdx.x * 16;

    // Fill: thread t -> row t>>4, 8 consecutive cols; two float4 global
    // loads, converted, one 16B LDS vector store.
    {
        int r  = tid >> 4;
        int c0 = (tid & 15) * 8;
        int row = row0 + r;
        if (row >= rows) row = rows - 1;        // clamp, no lane masking
        const float* xp = X + (size_t)row * HID + c0;
        float4 f0 = *(const float4*)(xp);
        float4 f1 = *(const float4*)(xp + 4);
        v8bf vb;
        vb[0] = (__bf16)f0.x; vb[1] = (__bf16)f0.y;
        vb[2] = (__bf16)f0.z; vb[3] = (__bf16)f0.w;
        vb[4] = (__bf16)f1.x; vb[5] = (__bf16)f1.y;
        vb[6] = (__bf16)f1.z; vb[7] = (__bf16)f1.w;
        *(v8bf*)&Xs[r][c0] = vb;
    }
    __syncthreads();

    const int wave     = tid >> 5;       // 0..7 -> output column tile
    const int lane     = tid & 31;       // wave32
    const int col_base = wave * 16;
    const int lhalf    = lane >> 4;      // 0 or 1
    const int lmod     = lane & 15;

    v8f acc = {};
    #pragma unroll
    for (int kk = 0; kk < HID; kk += 32) {
        // A fragment, row M = lmod (contiguous k = 16*lhalf + s, matching B).
        const int ab = kk + lhalf * 16;
        v16bf a;
        #pragma unroll
        for (int j = 0; j < 8; ++j) a[j]     = Xs[lmod][ab + j];
        #pragma unroll
        for (int j = 0; j < 8; ++j) a[8 + j] = Xs[lmod][ab + 8 + j];

        // B fragment, column N = lmod -> weight row (col_base + lmod).
        const __bf16* wr = Wbf + (size_t)(col_base + lmod) * HID + kk + lhalf * 16;
        v16bf b;
        #pragma unroll
        for (int j = 0; j < 16; ++j) b[j] = wr[j];

        acc = __builtin_amdgcn_wmma_f32_16x16x32_bf16(
            false, a, false, b, (short)0, acc, false, false);
    }

    const int   col = col_base + lmod;
    const float bv  = bias[col];

    if (row0 + 16 <= rows) {
        // Full tile: wave-uniform scalar branch, straight-line stores.
        #pragma unroll
        for (int r = 0; r < 8; ++r) {
            int row = row0 + r + (lhalf << 3);   // ISA 32-bit C layout
            float v = acc[r] + bv;
            if (RELU) v = fmaxf(v, 0.0f);
            if (HAS_RES) v += res[(size_t)row * HID + col];
            Y[(size_t)row * HID + col] = v;
        }
    } else {
        #pragma unroll
        for (int r = 0; r < 8; ++r) {
            int row = row0 + r + (lhalf << 3);
            if (row < rows) {
                float v = acc[r] + bv;
                if (RELU) v = fmaxf(v, 0.0f);
                if (HAS_RES) v += res[(size_t)row * HID + col];
                Y[(size_t)row * HID + col] = v;
            }
        }
    }
}

// ---------------------------------------------------------------------------
// Zero ft, init zmax=-inf, denom=0.
// ---------------------------------------------------------------------------
__global__ void init_kernel(float* __restrict__ zmax, float* __restrict__ denom,
                            float* __restrict__ ft, int n)
{
    int idx = blockIdx.x * blockDim.x + threadIdx.x;
    if (idx < n * HID) ft[idx] = 0.0f;
    if (idx < n * NHEAD) { zmax[idx] = -INFINITY; denom[idx] = 0.0f; }
}

// ---------------------------------------------------------------------------
// Edge pass 1: one wave32 per edge. Lane l loads float4 at offset 4*l of the
// gathered k[src]/q[dst] rows (512B fully-coalesced per row). Per-head dot
// (head = l>>2) reduced with two shfl_xor steps inside each 4-lane quad.
// z = dot + dist_decay stored; segment max over dst via float atomic max.
// ---------------------------------------------------------------------------
__global__ __launch_bounds__(256) void edge_z_kernel(
    const float* __restrict__ k, const float* __restrict__ q,
    const float* __restrict__ attn, const float* __restrict__ dd,
    const int* __restrict__ src, const int* __restrict__ dst,
    float* __restrict__ z, float* __restrict__ zmax, int E)
{
    int wid  = blockIdx.x * (blockDim.x >> 5) + (threadIdx.x >> 5);  // edge
    if (wid >= E) return;
    int lane = threadIdx.x & 31;
    int h    = lane >> 2;
    int sub  = lane & 3;

    int s = src[wid], d = dst[wid];
    float4 kv = *(const float4*)(k + (size_t)s * HID + lane * 4);
    float4 qv = *(const float4*)(q + (size_t)d * HID + lane * 4);
    float4 av = *(const float4*)(attn + h * DHEAD + sub * 4);

    float acc = lrelu(kv.x + qv.x) * av.x
              + lrelu(kv.y + qv.y) * av.y
              + lrelu(kv.z + qv.z) * av.z
              + lrelu(kv.w + qv.w) * av.w;
    acc += __shfl_xor(acc, 1, 32);
    acc += __shfl_xor(acc, 2, 32);

    if (sub == 0) {
        float zz = acc + dd[wid];
        z[(size_t)wid * NHEAD + h] = zz;
        atomic_max_float(&zmax[d * NHEAD + h], zz);
    }
}

// ---------------------------------------------------------------------------
// Edge pass 2: one wave32 per edge. ez = exp(z - zmax[dst]); denom add from
// quad leader; UNNORMALIZED message ez * bond scattered into ft with four
// coalesced global_atomic_add_f32 per lane (division by denom deferred).
// ---------------------------------------------------------------------------
__global__ __launch_bounds__(256) void edge_scatter_kernel(
    const float* __restrict__ z, const float* __restrict__ zmax,
    const float* __restrict__ bond, const int* __restrict__ dst,
    float* __restrict__ denom, float* __restrict__ ft, int E)
{
    int wid  = blockIdx.x * (blockDim.x >> 5) + (threadIdx.x >> 5);  // edge
    if (wid >= E) return;
    int lane = threadIdx.x & 31;
    int h    = lane >> 2;
    int sub  = lane & 3;

    int d = dst[wid];
    float zz = z[(size_t)wid * NHEAD + h];           // quad-broadcast read
    float ez = expf(zz - zmax[d * NHEAD + h]);
    if (sub == 0) atomicAdd(&denom[d * NHEAD + h], ez);

    float4 bv = *(const float4*)(bond + (size_t)wid * HID + lane * 4);
    float* fp = ft + (size_t)d * HID + lane * 4;
    atomicAdd(fp + 0, ez * bv.x);
    atomicAdd(fp + 1, ez * bv.y);
    atomicAdd(fp + 2, ez * bv.z);
    atomicAdd(fp + 3, ez * bv.w);
}

// ---------------------------------------------------------------------------
// ft[n, h*16+d] /= denom[n, h]
// ---------------------------------------------------------------------------
__global__ void normalize_kernel(float* __restrict__ ft,
                                 const float* __restrict__ denom, int n)
{
    int idx = blockIdx.x * blockDim.x + threadIdx.x;
    if (idx >= n * HID) return;
    int node = idx >> 7;
    int h    = (idx >> 4) & 7;
    ft[idx] /= denom[node * NHEAD + h];
}

// ---------------------------------------------------------------------------
static inline void launch_gemm(const float* X, const __bf16* W, const float* b,
                               const float* res, float* Y, int rows, bool relu,
                               int grid, hipStream_t s)
{
    if (res) {
        if (relu) gemm_bias_act_kernel<true,  true ><<<grid, 256, 0, s>>>(X, W, b, res, Y, rows);
        else      gemm_bias_act_kernel<false, true ><<<grid, 256, 0, s>>>(X, W, b, res, Y, rows);
    } else {
        if (relu) gemm_bias_act_kernel<true,  false><<<grid, 256, 0, s>>>(X, W, b, nullptr, Y, rows);
        else      gemm_bias_act_kernel<false, false><<<grid, 256, 0, s>>>(X, W, b, nullptr, Y, rows);
    }
}

extern "C" void kernel_launch(void* const* d_in, const int* in_sizes, int n_in,
                              void* d_out, int out_size, void* d_ws, size_t ws_size,
                              hipStream_t stream)
{
    const int*   src    = (const int*)  d_in[0];
    const int*   dst    = (const int*)  d_in[1];
    const float* bond   = (const float*)d_in[2];
    const float* atom   = (const float*)d_in[3];
    const float* dd     = (const float*)d_in[4];
    const float* k_b    = (const float*)d_in[6];
    const float* q_b    = (const float*)d_in[8];
    const float* attn   = (const float*)d_in[9];
    const float* lin1_b = (const float*)d_in[11];
    const float* lin2_b = (const float*)d_in[13];
    const float* r1a_b  = (const float*)d_in[15];
    const float* r1b_b  = (const float*)d_in[17];
    const float* r2a_b  = (const float*)d_in[19];
    const float* r2b_b  = (const float*)d_in[21];

    const int E = in_sizes[0];
    const int N = in_sizes[3] / HID;

    float* ws    = (float*)d_ws;
    float* k     = ws;
    float* q     = k     + (size_t)N * HID;
    float* ft    = q     + (size_t)N * HID;
    float* tmp   = ft    + (size_t)N * HID;
    float* z     = tmp   + (size_t)N * HID;
    float* zmax  = z     + (size_t)E * NHEAD;
    float* denom = zmax  + (size_t)N * NHEAD;
    __bf16* wbf  = (__bf16*)(denom + (size_t)N * NHEAD);
    const size_t WSZ = (size_t)HID * HID;             // 16384 elems / matrix

    float* he = (float*)d_out;

    const dim3 blk(256);
    const int gemm_grid = (N + 15) / 16;
    const int node_grid = (N * HID + 255) / 256;
    const int edge_grid = (E + 7) / 8;                // wave-per-edge
    const int conv_grid = (int)(WSZ / 256);

    // One-time bf16 weight conversion (order: kw,qw,lin1,lin2,r1a,r1b,r2a,r2b)
    const int w_idx[8] = {5, 7, 10, 12, 14, 16, 18, 20};
    for (int i = 0; i < 8; ++i)
        convert_w_kernel<<<conv_grid, blk, 0, stream>>>(
            (const float*)d_in[w_idx[i]], wbf + (size_t)i * WSZ);
    const __bf16* kwb   = wbf + 0 * WSZ;
    const __bf16* qwb   = wbf + 1 * WSZ;
    const __bf16* l1wb  = wbf + 2 * WSZ;
    const __bf16* l2wb  = wbf + 3 * WSZ;
    const __bf16* r1awb = wbf + 4 * WSZ;
    const __bf16* r1bwb = wbf + 5 * WSZ;
    const __bf16* r2awb = wbf + 6 * WSZ;
    const __bf16* r2bwb = wbf + 7 * WSZ;

    init_kernel<<<node_grid, blk, 0, stream>>>(zmax, denom, ft, N);

    launch_gemm(atom, kwb, k_b, nullptr, k, N, false, gemm_grid, stream);
    launch_gemm(atom, qwb, q_b, nullptr, q, N, false, gemm_grid, stream);

    edge_z_kernel<<<edge_grid, blk, 0, stream>>>(k, q, attn, dd, src, dst, z, zmax, E);
    edge_scatter_kernel<<<edge_grid, blk, 0, stream>>>(z, zmax, bond, dst, denom, ft, E);
    normalize_kernel<<<node_grid, blk, 0, stream>>>(ft, denom, N);

    // he = lin2(relu(lin1(ft))) + atom
    launch_gemm(ft,  l1wb, lin1_b, nullptr, tmp, N, true,  gemm_grid, stream);
    launch_gemm(tmp, l2wb, lin2_b, atom,    he,  N, false, gemm_grid, stream);
    // he += relu(r1b(relu(r1a(he))))
    launch_gemm(he,  r1awb, r1a_b, nullptr, tmp, N, true, gemm_grid, stream);
    launch_gemm(tmp, r1bwb, r1b_b, he,      he,  N, true, gemm_grid, stream);
    // he += relu(r2b(relu(r2a(he))))
    launch_gemm(he,  r2awb, r2a_b, nullptr, tmp, N, true, gemm_grid, stream);
    launch_gemm(tmp, r2bwb, r2b_b, he,      he,  N, true, gemm_grid, stream);
}
